// TimeAttention_19464791786088
// MI455X (gfx1250) — compile-verified, matrix operand order checked by reading, not run on hardware
//
#include <hip/hip_runtime.h>

typedef __bf16 bf16;
typedef __attribute__((ext_vector_type(16))) __bf16 v16bf;
typedef __attribute__((ext_vector_type(8)))  float  v8f;

namespace {
constexpr int kD    = 1024;   // d_model
constexpr int kH    = 16;     // query heads
constexpr int kHKV  = 4;      // kv heads
constexpr int kHD   = 64;     // head dim
constexpr int kT    = 64;     // time
constexpr int kS    = 256;    // spatial
constexpr int kB    = 2;      // batch
constexpr int kSeq  = kB * kS;            // 512 folded sequences
constexpr int kRows = kSeq * kT;          // 32768 token rows
constexpr int kKV   = kHKV * kHD;         // 256
constexpr int kQKV  = kD + 2 * kKV;       // 1536 fused qkv width
}

// ---------------------------------------------------------------------------
// WMMA helpers (gfx1250 wave32, v_wmma_f32_16x16x32_bf16)
// ---------------------------------------------------------------------------
__device__ __forceinline__ v8f wmma_bf16(v16bf a, v16bf b, v8f c) {
  return __builtin_amdgcn_wmma_f32_16x16x32_bf16(
      /*neg_a=*/false, a, /*neg_b=*/false, b,
      /*c_mod=*/(short)0, c, /*reuse_a=*/false, /*reuse_b=*/false);
}

// A fragment: 16x32 (MxK) tile, row-major, leading dim ld.
// lane 0..15: M=lane, K={0..7,16..23}; lane 16..31: M=lane-16, K={8..15,24..31}.
// Two contiguous 16B reads per lane -> ds_load_b128 pairs.
__device__ __forceinline__ v16bf frag_a(const bf16* tile, int ld) {
  const int lane = threadIdx.x & 31;
  const int m = lane & 15, kh = lane >> 4;
  const bf16* p0 = tile + m * ld + kh * 8;
  const bf16* p1 = p0 + 16;
  v16bf a;
#pragma unroll
  for (int e = 0; e < 8; ++e) a[e] = p0[e];
#pragma unroll
  for (int e = 0; e < 8; ++e) a[8 + e] = p1[e];
  return a;
}

// B fragment from a K-major tile: logical B[k][n] = tile[n*ld + k]
// (tile stored as N rows x K cols). Same contiguous access pattern as frag_a.
__device__ __forceinline__ v16bf frag_b_kmaj(const bf16* tile, int ld) {
  const int lane = threadIdx.x & 31;
  const int n = lane & 15, kh = lane >> 4;
  const bf16* p0 = tile + n * ld + kh * 8;
  const bf16* p1 = p0 + 16;
  v16bf b;
#pragma unroll
  for (int e = 0; e < 8; ++e) b[e] = p0[e];
#pragma unroll
  for (int e = 0; e < 8; ++e) b[8 + e] = p1[e];
  return b;
}

// ---------------------------------------------------------------------------
// Kernel 1: x (B,T,S,D) f32 -> xt (B*S*T, D) bf16  (transpose fused in cast)
// ---------------------------------------------------------------------------
__global__ __launch_bounds__(256) void cvt_x_kernel(const float* __restrict__ x,
                                                    bf16* __restrict__ xt) {
  const size_t i4 = ((size_t)blockIdx.x * blockDim.x + threadIdx.x) * 4;
  if (i4 >= (size_t)kRows * kD) return;
  const int d   = (int)(i4 & (kD - 1));
  const int row = (int)(i4 >> 10);
  const int t = row & (kT - 1);
  const int nn = row >> 6;            // b*S + s
  const int b = nn >> 8, s = nn & (kS - 1);
  const float4 v = *(const float4*)(x + (((size_t)((b * kT + t) * kS + s)) << 10) + d);
  __align__(8) bf16 o[4] = {(bf16)v.x, (bf16)v.y, (bf16)v.z, (bf16)v.w};
  *(uint2*)(xt + i4) = *(const uint2*)o;
}

// ---------------------------------------------------------------------------
// Kernel 2: build TRANSPOSED bf16 weights (K-contiguous rows):
//   wqkvT: 1536 x 1024, woT: 1024 x 1024
// ---------------------------------------------------------------------------
__global__ __launch_bounds__(256) void cvt_w_kernel(const float* __restrict__ wq,
                                                    const float* __restrict__ wk,
                                                    const float* __restrict__ wv,
                                                    const float* __restrict__ wo,
                                                    bf16* __restrict__ wqkvT,
                                                    bf16* __restrict__ woT) {
  const int i = blockIdx.x * 256 + threadIdx.x;
  const int nQKV = kD * kQKV;
  if (i < nQKV) {
    const int col = i >> 10;          // 0..1535
    const int k   = i & (kD - 1);
    float v;
    if (col < kD)            v = wq[k * kD + col];
    else if (col < kD + kKV) v = wk[k * kKV + (col - kD)];
    else                     v = wv[k * kKV + (col - kD - kKV)];
    wqkvT[i] = (bf16)v;
  } else {
    const int i2 = i - nQKV;
    if (i2 < kD * kD) {
      const int col = i2 >> 10;
      const int k   = i2 & (kD - 1);
      woT[i2] = (bf16)wo[k * kD + col];
    }
  }
}

// ---------------------------------------------------------------------------
// Kernel 3: fused QKV GEMM (bf16 WMMA, 128x64 block, 2x2 wave tiles,
// double-buffered LDS) + RMSNorm + RoPE epilogue.
// block = 256 thr (8 waves = 4 row-groups x 2 col-groups of 32x32).
// grid = (kRows/128, kQKV/64). V is written TRANSPOSED per (seq, kv-head).
// ---------------------------------------------------------------------------
__global__ __launch_bounds__(256) void qkv_gemm_kernel(
    const bf16* __restrict__ X, const bf16* __restrict__ WT,
    const float* __restrict__ q_scale, const float* __restrict__ k_scale,
    const float* __restrict__ rope_cos, const float* __restrict__ rope_sin,
    bf16* __restrict__ qbuf, bf16* __restrict__ kbuf, bf16* __restrict__ vt) {
  __shared__ __align__(16) bf16 sA[2][128 * 32];  // M x K
  __shared__ __align__(16) bf16 sBt[2][64 * 32];  // N x K (K-major)
  __shared__ float sC[128 * 64];

  const int tid = threadIdx.x;
  const int wave = tid >> 5, lane = tid & 31;
  const int row0 = blockIdx.x * 128;
  const int col0 = blockIdx.y * 64;
  const int rg = wave & 3;            // 32-row group
  const int cg = wave >> 2;           // 32-col group

  v8f acc00 = {}, acc01 = {}, acc10 = {}, acc11 = {};

  const int a0r = tid >> 2, a0c = (tid & 3) * 8;  // A rows 0..63
  const int a1r = a0r + 64;                       // A rows 64..127
  const int br  = tid >> 2, bc  = (tid & 3) * 8;  // B 64x32

  // preload K-step 0
  {
    *(uint4*)&sA[0][a0r * 32 + a0c] = *(const uint4*)(X  + (size_t)(row0 + a0r) * kD + a0c);
    *(uint4*)&sA[0][a1r * 32 + a0c] = *(const uint4*)(X  + (size_t)(row0 + a1r) * kD + a0c);
    *(uint4*)&sBt[0][br * 32 + bc]  = *(const uint4*)(WT + (size_t)(col0 + br) * kD + bc);
  }
  __syncthreads();

  for (int k0 = 0; k0 < kD; k0 += 32) {
    const int cur = (k0 >> 5) & 1;
    const bool more = (k0 + 32) < kD;
    uint4 na0 = {}, na1 = {}, nb = {};
    if (more) {   // issue next tile's loads before the matrix ops
      na0 = *(const uint4*)(X  + (size_t)(row0 + a0r) * kD + (k0 + 32) + a0c);
      na1 = *(const uint4*)(X  + (size_t)(row0 + a1r) * kD + (k0 + 32) + a0c);
      nb  = *(const uint4*)(WT + (size_t)(col0 + br)  * kD + (k0 + 32) + bc);
      if (k0 + 64 < kD) {
        __builtin_prefetch(X  + (size_t)(row0 + a0r) * kD + (k0 + 64) + a0c, 0, 1);
        __builtin_prefetch(X  + (size_t)(row0 + a1r) * kD + (k0 + 64) + a0c, 0, 1);
        __builtin_prefetch(WT + (size_t)(col0 + br)  * kD + (k0 + 64) + bc, 0, 1);
      }
    }
    const v16bf a0 = frag_a(&sA[cur][(rg * 32) * 32], 32);
    const v16bf a1 = frag_a(&sA[cur][(rg * 32 + 16) * 32], 32);
    const v16bf b0 = frag_b_kmaj(&sBt[cur][(cg * 32) * 32], 32);
    const v16bf b1 = frag_b_kmaj(&sBt[cur][(cg * 32 + 16) * 32], 32);
    acc00 = wmma_bf16(a0, b0, acc00);
    acc01 = wmma_bf16(a0, b1, acc01);
    acc10 = wmma_bf16(a1, b0, acc10);
    acc11 = wmma_bf16(a1, b1, acc11);
    if (more) {
      *(uint4*)&sA[cur ^ 1][a0r * 32 + a0c] = na0;
      *(uint4*)&sA[cur ^ 1][a1r * 32 + a0c] = na1;
      *(uint4*)&sBt[cur ^ 1][br * 32 + bc]  = nb;
    }
    __syncthreads();
  }

  // spill accumulators (C layout: M = vgpr + 8*(lane>=16), N = lane&15)
  {
    const int rb0 = rg * 32 + ((lane >> 4) << 3);
    const int rb1 = rb0 + 16;
    const int c0 = cg * 32 + (lane & 15);
    const int c1 = c0 + 16;
#pragma unroll
    for (int r = 0; r < 8; ++r) {
      sC[(rb0 + r) * 64 + c0] = acc00[r];
      sC[(rb0 + r) * 64 + c1] = acc01[r];
      sC[(rb1 + r) * 64 + c0] = acc10[r];
      sC[(rb1 + r) * 64 + c1] = acc11[r];
    }
  }
  __syncthreads();

  // epilogue (64-wide tile == exactly one head; 128 rows per block)
  if (tid < 128) {
    if (col0 < kD) {                      // ---- Q: RMSNorm + RoPE ----
      const int r = tid;
      const int grow = row0 + r;
      const int t = grow & (kT - 1);
      float ssum = 0.f;
#pragma unroll
      for (int j = 0; j < 64; ++j) { const float v = sC[r * 64 + j]; ssum += v * v; }
      const float rms = rsqrtf(ssum * (1.0f / 64.0f) + 1e-6f);
      bf16* dst = qbuf + (size_t)grow * kD + col0;
#pragma unroll
      for (int j = 0; j < 32; ++j) {
        const float u1 = sC[r * 64 + j]      * rms * q_scale[j];
        const float u2 = sC[r * 64 + j + 32] * rms * q_scale[j + 32];
        const float c = rope_cos[t * 32 + j], s = rope_sin[t * 32 + j];
        dst[j]      = (bf16)(u1 * c - u2 * s);
        dst[j + 32] = (bf16)(u1 * s + u2 * c);
      }
    } else if (col0 < kD + kKV) {         // ---- K: RMSNorm + RoPE ----
      const int r = tid;
      const int grow = row0 + r;
      const int t = grow & (kT - 1);
      float ssum = 0.f;
#pragma unroll
      for (int j = 0; j < 64; ++j) { const float v = sC[r * 64 + j]; ssum += v * v; }
      const float rms = rsqrtf(ssum * (1.0f / 64.0f) + 1e-6f);
      bf16* dst = kbuf + (size_t)grow * kKV + (col0 - kD);
#pragma unroll
      for (int j = 0; j < 32; ++j) {
        const float u1 = sC[r * 64 + j]      * rms * k_scale[j];
        const float u2 = sC[r * 64 + j + 32] * rms * k_scale[j + 32];
        const float c = rope_cos[t * 32 + j], s = rope_sin[t * 32 + j];
        dst[j]      = (bf16)(u1 * c - u2 * s);
        dst[j + 32] = (bf16)(u1 * s + u2 * c);
      }
    } else {                              // ---- V: store TRANSPOSED (hd, t) ----
      // block covers two whole sequences (row0 is 128-aligned, T=64)
      const int half = tid >> 6;          // which sequence half of the block
      const int j = tid & 63;             // head-dim index
      const int nseq = (row0 >> 6) + half;
      const int kvh = (col0 - kD - kKV) >> 6;
      bf16* dst = vt + (((size_t)(nseq * kHKV + kvh) * kHD + j) << 6);
#pragma unroll
      for (int t2 = 0; t2 < 64; ++t2) dst[t2] = (bf16)sC[(half * 64 + t2) * 64 + j];
    }
  }
}

// ---------------------------------------------------------------------------
// Kernel 4: attention per (sequence, head); WMMA for Q*K^T and P*V.
// grid = (kSeq, kH), block = 256 (8 waves). V arrives transposed (HD x T).
// ---------------------------------------------------------------------------
__global__ __launch_bounds__(256) void attn_kernel(const bf16* __restrict__ qbuf,
                                                   const bf16* __restrict__ kbuf,
                                                   const bf16* __restrict__ vt,
                                                   bf16* __restrict__ obuf) {
  __shared__ __align__(16) bf16 sQ[64 * 64];   // t x hd
  __shared__ __align__(16) bf16 sK[64 * 64];   // s x hd  (K-major for B frag)
  __shared__ __align__(16) bf16 sVt[64 * 64];  // hd x s  (K-major for B frag)
  __shared__ __align__(16) bf16 sP[64 * 64];   // t x s
  __shared__ float sS[64 * 64];

  const int n = blockIdx.x, h = blockIdx.y, kvh = h >> 2;
  const int tid = threadIdx.x;
  const int wave = tid >> 5, lane = tid & 31;
  const int mi = wave & 3, nj = (wave >> 2) * 2;

  const bf16* vbase = vt + ((size_t)(n * kHKV + kvh) * kHD) * 64;
#pragma unroll
  for (int it = 0; it < 2; ++it) {
    const int i = tid * 8 + it * 2048;
    const int r = i >> 6, c = i & 63;
    *(uint4*)&sQ[i]  = *(const uint4*)(qbuf + (size_t)(n * kT + r) * kD  + h   * kHD + c);
    *(uint4*)&sK[i]  = *(const uint4*)(kbuf + (size_t)(n * kT + r) * kKV + kvh * kHD + c);
    *(uint4*)&sVt[i] = *(const uint4*)(vbase + i);
  }
  __syncthreads();

  // S = Q * K^T   (B[k][n] = K[n][k], sK is K-major per key row)
  v8f s0 = {}, s1 = {};
#pragma unroll
  for (int k0 = 0; k0 < kHD; k0 += 32) {
    const v16bf a  = frag_a(&sQ[mi * 16 * 64 + k0], 64);
    const v16bf b0 = frag_b_kmaj(&sK[(nj * 16) * 64 + k0], 64);
    const v16bf b1 = frag_b_kmaj(&sK[((nj + 1) * 16) * 64 + k0], 64);
    s0 = wmma_bf16(a, b0, s0);
    s1 = wmma_bf16(a, b1, s1);
  }

  // scale + softcap + causal mask -> LDS
  {
    const int rbase = mi * 16 + ((lane >> 4) << 3);
    const int c0 = nj * 16 + (lane & 15);
    const int c1 = (nj + 1) * 16 + (lane & 15);
#pragma unroll
    for (int r = 0; r < 8; ++r) {
      const int R = rbase + r;
      float v0 = 50.0f * tanhf(s0[r] * 0.125f * 0.02f);   // (s/8)/50
      float v1 = 50.0f * tanhf(s1[r] * 0.125f * 0.02f);
      if (c0 > R) v0 = -1e30f;
      if (c1 > R) v1 = -1e30f;
      sS[R * 64 + c0] = v0;
      sS[R * 64 + c1] = v1;
    }
  }
  __syncthreads();

  // row softmax -> bf16 probabilities
  if (tid < 64) {
    float mx = -1e30f;
#pragma unroll
    for (int j = 0; j < 64; ++j) mx = fmaxf(mx, sS[tid * 64 + j]);
    float sum = 0.f;
#pragma unroll
    for (int j = 0; j < 64; ++j) {
      const float e = __expf(sS[tid * 64 + j] - mx);
      sS[tid * 64 + j] = e;
      sum += e;
    }
    const float inv = 1.0f / sum;
#pragma unroll
    for (int j = 0; j < 64; ++j) sP[tid * 64 + j] = (bf16)(sS[tid * 64 + j] * inv);
  }
  __syncthreads();

  // O = P * V   (B[k][n] = V[k][n] = sVt[n*64 + k], K-major per hd row)
  v8f o0 = {}, o1 = {};
#pragma unroll
  for (int k0 = 0; k0 < kT; k0 += 32) {
    const v16bf a  = frag_a(&sP[mi * 16 * 64 + k0], 64);
    const v16bf b0 = frag_b_kmaj(&sVt[(nj * 16) * 64 + k0], 64);
    const v16bf b1 = frag_b_kmaj(&sVt[((nj + 1) * 16) * 64 + k0], 64);
    o0 = wmma_bf16(a, b0, o0);
    o1 = wmma_bf16(a, b1, o1);
  }

  // store O as bf16 for the output projection
  {
    const int rbase = mi * 16 + ((lane >> 4) << 3);
    const int c0 = nj * 16 + (lane & 15);
    const int c1 = (nj + 1) * 16 + (lane & 15);
    bf16* base = obuf + (size_t)(n * kT) * kD + h * kHD;
#pragma unroll
    for (int r = 0; r < 8; ++r) {
      base[(size_t)(rbase + r) * kD + c0] = (bf16)o0[r];
      base[(size_t)(rbase + r) * kD + c1] = (bf16)o1[r];
    }
  }
}

// ---------------------------------------------------------------------------
// Kernel 5: output projection (bf16 WMMA, 128x64 block, 2x2 wave tiles,
// double-buffered), fp32 scatter store. grid = (kRows/128, kD/64)
// ---------------------------------------------------------------------------
__global__ __launch_bounds__(256) void out_gemm_kernel(const bf16* __restrict__ A,
                                                       const bf16* __restrict__ WT,
                                                       float* __restrict__ out) {
  __shared__ __align__(16) bf16 sA[2][128 * 32];
  __shared__ __align__(16) bf16 sBt[2][64 * 32];

  const int tid = threadIdx.x;
  const int wave = tid >> 5, lane = tid & 31;
  const int row0 = blockIdx.x * 128;
  const int col0 = blockIdx.y * 64;
  const int rg = wave & 3, cg = wave >> 2;

  v8f acc00 = {}, acc01 = {}, acc10 = {}, acc11 = {};
  const int a0r = tid >> 2, a0c = (tid & 3) * 8;
  const int a1r = a0r + 64;
  const int br  = tid >> 2, bc  = (tid & 3) * 8;

  {
    *(uint4*)&sA[0][a0r * 32 + a0c] = *(const uint4*)(A  + (size_t)(row0 + a0r) * kD + a0c);
    *(uint4*)&sA[0][a1r * 32 + a0c] = *(const uint4*)(A  + (size_t)(row0 + a1r) * kD + a0c);
    *(uint4*)&sBt[0][br * 32 + bc]  = *(const uint4*)(WT + (size_t)(col0 + br) * kD + bc);
  }
  __syncthreads();

  for (int k0 = 0; k0 < kD; k0 += 32) {
    const int cur = (k0 >> 5) & 1;
    const bool more = (k0 + 32) < kD;
    uint4 na0 = {}, na1 = {}, nb = {};
    if (more) {
      na0 = *(const uint4*)(A  + (size_t)(row0 + a0r) * kD + (k0 + 32) + a0c);
      na1 = *(const uint4*)(A  + (size_t)(row0 + a1r) * kD + (k0 + 32) + a0c);
      nb  = *(const uint4*)(WT + (size_t)(col0 + br)  * kD + (k0 + 32) + bc);
      if (k0 + 64 < kD) {
        __builtin_prefetch(A  + (size_t)(row0 + a0r) * kD + (k0 + 64) + a0c, 0, 1);
        __builtin_prefetch(A  + (size_t)(row0 + a1r) * kD + (k0 + 64) + a0c, 0, 1);
        __builtin_prefetch(WT + (size_t)(col0 + br)  * kD + (k0 + 64) + bc, 0, 1);
      }
    }
    const v16bf a0 = frag_a(&sA[cur][(rg * 32) * 32], 32);
    const v16bf a1 = frag_a(&sA[cur][(rg * 32 + 16) * 32], 32);
    const v16bf b0 = frag_b_kmaj(&sBt[cur][(cg * 32) * 32], 32);
    const v16bf b1 = frag_b_kmaj(&sBt[cur][(cg * 32 + 16) * 32], 32);
    acc00 = wmma_bf16(a0, b0, acc00);
    acc01 = wmma_bf16(a0, b1, acc01);
    acc10 = wmma_bf16(a1, b0, acc10);
    acc11 = wmma_bf16(a1, b1, acc11);
    if (more) {
      *(uint4*)&sA[cur ^ 1][a0r * 32 + a0c] = na0;
      *(uint4*)&sA[cur ^ 1][a1r * 32 + a0c] = na1;
      *(uint4*)&sBt[cur ^ 1][br * 32 + bc]  = nb;
    }
    __syncthreads();
  }

  // fp32 scatter store: row (b*S+s, t) -> out[(b,t,s,:)]
  const int rb0 = row0 + rg * 32 + ((lane >> 4) << 3);
  const int c0 = col0 + cg * 32 + (lane & 15);
  const int c1 = c0 + 16;
#pragma unroll
  for (int r = 0; r < 8; ++r) {
#pragma unroll
    for (int half = 0; half < 2; ++half) {
      const int R = rb0 + half * 16 + r;
      const int t = R & (kT - 1);
      const int nn = R >> 6;
      const int b = nn >> 8, s = nn & (kS - 1);
      const size_t off = ((size_t)((b * kT + t) * kS + s)) << 10;
      out[off + c0] = half ? acc10[r] : acc00[r];
      out[off + c1] = half ? acc11[r] : acc01[r];
    }
  }
}

// ---------------------------------------------------------------------------
// Host launcher
// ---------------------------------------------------------------------------
extern "C" void kernel_launch(void* const* d_in, const int* in_sizes, int n_in,
                              void* d_out, int out_size, void* d_ws, size_t ws_size,
                              hipStream_t stream) {
  (void)in_sizes; (void)n_in; (void)out_size; (void)ws_size;
  const float* x        = (const float*)d_in[0];
  const float* wq       = (const float*)d_in[1];
  const float* wk       = (const float*)d_in[2];
  const float* wv       = (const float*)d_in[3];
  const float* wo       = (const float*)d_in[4];
  const float* q_scale  = (const float*)d_in[5];
  const float* k_scale  = (const float*)d_in[6];
  const float* rope_cos = (const float*)d_in[7];
  const float* rope_sin = (const float*)d_in[8];

  char* p = (char*)d_ws;
  auto take = [&](size_t bytes) -> void* {
    void* r = (void*)p;
    p += (bytes + 255) & ~(size_t)255;
    return r;
  };
  bf16* xt    = (bf16*)take((size_t)kRows * kD * sizeof(bf16));
  bf16* wqkvT = (bf16*)take((size_t)kD * kQKV * sizeof(bf16));
  bf16* woT   = (bf16*)take((size_t)kD * kD * sizeof(bf16));
  bf16* qb    = (bf16*)take((size_t)kRows * kD * sizeof(bf16));
  bf16* kb    = (bf16*)take((size_t)kRows * kKV * sizeof(bf16));
  bf16* vtb   = (bf16*)take((size_t)kRows * kKV * sizeof(bf16));
  bf16* ob    = (bf16*)take((size_t)kRows * kD * sizeof(bf16));

  cvt_x_kernel<<<(int)(((size_t)kRows * kD / 4) / 256), 256, 0, stream>>>(x, xt);
  cvt_w_kernel<<<(kD * kQKV + kD * kD) / 256, 256, 0, stream>>>(wq, wk, wv, wo, wqkvT, woT);
  qkv_gemm_kernel<<<dim3(kRows / 128, kQKV / 64), 256, 0, stream>>>(
      xt, wqkvT, q_scale, k_scale, rope_cos, rope_sin, qb, kb, vtb);
  attn_kernel<<<dim3(kSeq, kH), 256, 0, stream>>>(qb, kb, vtb, ob);
  out_gemm_kernel<<<dim3(kRows / 128, kD / 64), 256, 0, stream>>>(ob, woT, (float*)d_out);
}
